// SelfAttention_41850161332908
// MI455X (gfx1250) — matrix operand order repaired and checked by reading, not compiled, not run on hardware
//
#include <hip/hip_runtime.h>
#include <hip/hip_bf16.h>
#include <math.h>

// ---------------------------------------------------------------------------
// Self-attention on MI455X (gfx1250): all matmuls via v_wmma_f32_16x16x32_f16.
// Pipeline: cvt(x,wqkvT,woutT) -> GEMM(qkv split) -> flash-attn -> GEMM(out).
// GEMM LDS staging uses GLOBAL_LOAD_ASYNC_TO_LDS_B128 when the builtin exists.
// ---------------------------------------------------------------------------

#define B_  4
#define L_  2048
#define D_  1024
#define H_  16
#define KD_ 64

typedef __attribute__((ext_vector_type(16))) _Float16     v16h;
typedef __attribute__((ext_vector_type(8)))  float        v8f;
typedef __attribute__((ext_vector_type(4)))  _Float16     v4h;
typedef __attribute__((ext_vector_type(4)))  float        f32x4;
typedef __attribute__((ext_vector_type(4)))  unsigned int u32x4;
typedef __attribute__((ext_vector_type(4)))  int          i32x4;

union Frag { v16h v; u32x4 q[2]; };

static __device__ __forceinline__ v8f wmma_f16(v16h a, v16h b, v8f c) {
  // D = A(16x32 f16) * B(32x16 f16) + C(16x16 f32)
  return __builtin_amdgcn_wmma_f32_16x16x32_f16(false, a, false, b, (short)0, c,
                                                false, false);
}

// ---- async global->LDS (CDNA5 ASYNCcnt path), with safe fallback ----------
#if __has_builtin(__builtin_amdgcn_global_load_async_to_lds_b128)
#define ASYNC_LDS 1
typedef __attribute__((address_space(1))) i32x4 GI4;   // global int4
typedef __attribute__((address_space(3))) i32x4 LI4;   // LDS int4
static __device__ __forceinline__ GI4* as_global(const void* p) {
  return (GI4*)(unsigned long long)p;
}
static __device__ __forceinline__ LI4* as_lds(void* p) {
  // flat LDS address low 32 bits == wave-relative LDS offset
  return (LI4*)(unsigned int)(unsigned long long)p;
}
static __device__ __forceinline__ void async_copy16(const void* g, void* l) {
  __builtin_amdgcn_global_load_async_to_lds_b128(as_global(g), as_lds(l), 0, 0);
}
static __device__ __forceinline__ void wait_async() {
#if __has_builtin(__builtin_amdgcn_s_wait_asynccnt)
  __builtin_amdgcn_s_wait_asynccnt(0);
#else
  asm volatile("s_wait_asynccnt 0x0" ::: "memory");
#endif
}
#endif

// ---------------- conversion kernels ----------------

__global__ __launch_bounds__(256)
void cvt_f16_kernel(const float* __restrict__ in, _Float16* __restrict__ out, int n) {
  int i = (blockIdx.x * 256 + threadIdx.x) * 4;
  if (i + 3 < n) {
    f32x4 f = *(const f32x4*)&in[i];
    v4h h = { (_Float16)f[0], (_Float16)f[1], (_Float16)f[2], (_Float16)f[3] };
    *(v4h*)&out[i] = h;
  }
}

// in [1024][C] f32 (row-major)  ->  out [C][1024] f16 (transposed)
__global__ __launch_bounds__(256)
void cvt_transpose_kernel(const float* __restrict__ in, _Float16* __restrict__ out, int C) {
  const int i = blockIdx.x * 256 + threadIdx.x; // output index over C*1024
  const int r = i & 1023;                       // K index (rows of in)
  const int c = i >> 10;                        // N index (cols of in)
  out[i] = (_Float16)in[(size_t)r * C + c];
}

// ---------------- WMMA GEMM: C[M,N] = A[M,K] * Bt[N,K]^T ----------------
// Block tile 128x128, K-tile 32. 8 waves as 2(M)x4(N); each wave: 4x2 tiles.
// MODE 0: QKV-split f16 epilogue (Q/K row-major per head, V transposed).
// MODE 1: plain f32 store.

template<int MODE>
__global__ __launch_bounds__(256)
void gemm_f16_kernel(const _Float16* __restrict__ A, const _Float16* __restrict__ Bt,
                     int M, int N, int K,
                     _Float16* __restrict__ qO, _Float16* __restrict__ kO,
                     _Float16* __restrict__ vTO, float* __restrict__ fO)
{
  __shared__ _Float16 sA[128][40];   // +8 halves pad: distinct banks per row
  __shared__ _Float16 sB[128][40];

  const int nbx = N >> 7;
  const int bx = blockIdx.x % nbx;
  const int by = blockIdx.x / nbx;
  const int m0 = by << 7, n0 = bx << 7;
  const int tid  = threadIdx.x;
  const int lane = tid & 31, w = tid >> 5;
  const int wm = w >> 2, wn = w & 3;
  const int half = lane >> 4, l16 = lane & 15;

  v8f acc[4][2] = {};

  for (int k0 = 0; k0 < K; k0 += 32) {
    // cooperative tile staging: 512 x 16B chunks each for A and Bt
    #pragma unroll
    for (int it = 0; it < 2; ++it) {
      int c   = tid + (it << 8);
      int row = c >> 2, kc = (c & 3) << 3;
#ifdef ASYNC_LDS
      async_copy16(&A [(size_t)(m0 + row) * K + k0 + kc], &sA[row][kc]);
      async_copy16(&Bt[(size_t)(n0 + row) * K + k0 + kc], &sB[row][kc]);
#else
      *(u32x4*)&sA[row][kc] = *(const u32x4*)&A [(size_t)(m0 + row) * K + k0 + kc];
      *(u32x4*)&sB[row][kc] = *(const u32x4*)&Bt[(size_t)(n0 + row) * K + k0 + kc];
#endif
    }
    if (k0 + 32 < K) {   // global_prefetch_b8 of next K-tile
      __builtin_prefetch(&A [(size_t)(m0 + (tid & 127)) * K + k0 + 32], 0, 0);
      __builtin_prefetch(&Bt[(size_t)(n0 + (tid & 127)) * K + k0 + 32], 0, 0);
    }
#ifdef ASYNC_LDS
    wait_async();        // this wave's async LDS writes complete
#endif
    __syncthreads();     // all waves' tiles visible

    // B fragments: lane holds col n=l16, 16 contiguous K at half*16
    v16h bf[2];
    #pragma unroll
    for (int j = 0; j < 2; ++j) {
      Frag f;
      const int n = (wn << 5) + (j << 4) + l16;
      f.q[0] = *(const u32x4*)&sB[n][(half << 4)];
      f.q[1] = *(const u32x4*)&sB[n][(half << 4) + 8];
      bf[j] = f.v;
    }
    // A fragments: lane holds row m=l16, 16 contiguous K at half*16
    #pragma unroll
    for (int i = 0; i < 4; ++i) {
      Frag fa;
      const int mr = (wm << 6) + (i << 4) + l16;
      fa.q[0] = *(const u32x4*)&sA[mr][(half << 4)];
      fa.q[1] = *(const u32x4*)&sA[mr][(half << 4) + 8];
      #pragma unroll
      for (int j = 0; j < 2; ++j)
        acc[i][j] = wmma_f16(fa.v, bf[j], acc[i][j]);
    }
    __syncthreads();
  }

  // Epilogue. C layout: reg r of lane -> (m = r + 8*half, n = l16).
  // part/head are uniform per 16-wide tile; per-r strides are constant.
  #pragma unroll
  for (int i = 0; i < 4; ++i) {
    const int mg0 = m0 + (wm << 6) + (i << 4) + (half << 3);  // +r below
    #pragma unroll
    for (int j = 0; j < 2; ++j) {
      if (MODE == 0) {
        const int ngb  = n0 + (wn << 5) + (j << 4);  // scalar (wave-uniform)
        const int part = ngb >> 10;
        const int hh   = (ngb & 1023) >> 6;
        const int kd   = (ngb & 63) + l16;
        const int b    = mg0 >> 11, l0 = mg0 & 2047;
        const size_t bh = (size_t)(b * H_ + hh);
        if (part == 0) {
          _Float16* p = qO + (bh * L_ + l0) * KD_ + kd;
          #pragma unroll
          for (int r = 0; r < 8; ++r) p[r * KD_] = (_Float16)acc[i][j][r];
        } else if (part == 1) {
          _Float16* p = kO + (bh * L_ + l0) * KD_ + kd;
          #pragma unroll
          for (int r = 0; r < 8; ++r) p[r * KD_] = (_Float16)acc[i][j][r];
        } else {
          // V^T: r is the contiguous axis -> two packed 8B stores
          _Float16* p = vTO + (bh * KD_ + kd) * (size_t)L_ + l0;
          v4h h0 = { (_Float16)acc[i][j][0], (_Float16)acc[i][j][1],
                     (_Float16)acc[i][j][2], (_Float16)acc[i][j][3] };
          v4h h1 = { (_Float16)acc[i][j][4], (_Float16)acc[i][j][5],
                     (_Float16)acc[i][j][6], (_Float16)acc[i][j][7] };
          *(v4h*)&p[0] = h0;
          *(v4h*)&p[4] = h1;
        }
      } else {
        const int ng = n0 + (wn << 5) + (j << 4) + l16;
        float* p = fO + (size_t)mg0 * N + ng;
        #pragma unroll
        for (int r = 0; r < 8; ++r) p[(size_t)r * N] = acc[i][j][r];
      }
    }
  }
}

// ---------------- flash attention (transposed-score formulation) ------------
// Per wave: one 16-query tile. S^T = K(16x64) @ Q^T(64x16) so each lane owns a
// full query column -> softmax reduction = 8 regs + one shfl_xor(16).
// O^T(64x16) += V^T(16x32) @ P^T(32x16); P^T exchanged via wave-private LDS.

__global__ __launch_bounds__(256)
void attn_kernel(const _Float16* __restrict__ Q, const _Float16* __restrict__ Km,
                 const _Float16* __restrict__ Vt, const unsigned char* __restrict__ pad,
                 _Float16* __restrict__ O)
{
  __shared__ _Float16 sP[8][16][40];   // per-wave P^T exchange, [q][key], padded

  const int tid = threadIdx.x, lane = tid & 31, w = tid >> 5;
  const int half = lane >> 4, l16 = lane & 15;
  const int bh = blockIdx.x;                    // b*H + h
  const int b  = bh >> 4, h = bh & 15;
  const int q0 = (blockIdx.y << 7) + (w << 4);  // this wave's query tile
  const int q  = q0 + l16;                      // this lane's query column

  // Q^T B-fragments (two 32-wide kd chunks); lane: col q=l16, 16 contiguous kd
  v16h qf[2];
  {
    const _Float16* qrow = Q + ((size_t)bh * L_ + q) * KD_;
    #pragma unroll
    for (int c = 0; c < 2; ++c) {
      Frag f;
      f.q[0] = *(const u32x4*)&qrow[(c << 5) + (half << 4)];
      f.q[1] = *(const u32x4*)&qrow[(c << 5) + (half << 4) + 8];
      qf[c] = f.v;
    }
  }

  v8f o[4] = {};                       // O^T accum: 4 kd-tiles of 16
  float m_run = -1e30f, l_run = 0.f;
  const int nkb = (q0 + 16 + 31) >> 5; // causal: keys <= q0+15

  for (int kb = 0; kb < nkb; ++kb) {
    const int kbase = kb << 5;

    // S^T tiles (keys kt*16..kt*16+15 x 16 queries), k-dim = kd (2 x 32)
    v8f s[2];
    #pragma unroll
    for (int kt = 0; kt < 2; ++kt) {
      const _Float16* krow = Km + ((size_t)bh * L_ + kbase + (kt << 4) + l16) * KD_;
      v8f z = {};
      #pragma unroll
      for (int c = 0; c < 2; ++c) {
        Frag f;
        f.q[0] = *(const u32x4*)&krow[(c << 5) + (half << 4)];
        f.q[1] = *(const u32x4*)&krow[(c << 5) + (half << 4) + 8];
        z = wmma_f16(f.v, qf[c], z);
      }
      s[kt] = z;
    }

    // pad mask: this lane's 2x8 contiguous key bytes
    const unsigned long long pm0 =
        *(const unsigned long long*)&pad[(size_t)b * L_ + kbase + (half << 3)];
    const unsigned long long pm1 =
        *(const unsigned long long*)&pad[(size_t)b * L_ + kbase + 16 + (half << 3)];

    // scale + causal/pad mask + block max (per query column)
    float mb = -1e30f;
    #pragma unroll
    for (int kt = 0; kt < 2; ++kt) {
      #pragma unroll
      for (int r = 0; r < 8; ++r) {
        const int key = kbase + (kt << 4) + r + (half << 3);
        float v = s[kt][r] * 0.125f;   // 1/sqrt(64)
        const unsigned long long pm = kt ? pm1 : pm0;
        if (key > q || ((pm >> (r << 3)) & 0xffull)) v = -1e30f;
        s[kt][r] = v;
        mb = fmaxf(mb, v);
      }
    }
    mb = fmaxf(mb, __shfl_xor(mb, 16, 32));
    const float mnew  = fmaxf(m_run, mb);
    const float alpha = __expf(m_run - mnew);

    float bs = 0.f;
    #pragma unroll
    for (int kt = 0; kt < 2; ++kt) {
      float p[8];
      #pragma unroll
      for (int r = 0; r < 8; ++r) {
        p[r] = __expf(s[kt][r] - mnew);
        bs += p[r];
      }
      // keys are contiguous in r -> two packed 8B LDS stores
      v4h h0 = { (_Float16)p[0], (_Float16)p[1], (_Float16)p[2], (_Float16)p[3] };
      v4h h1 = { (_Float16)p[4], (_Float16)p[5], (_Float16)p[6], (_Float16)p[7] };
      _Float16* sp = &sP[w][l16][(kt << 4) + (half << 3)];
      *(v4h*)&sp[0] = h0;
      *(v4h*)&sp[4] = h1;
    }
    bs += __shfl_xor(bs, 16, 32);
    l_run = l_run * alpha + bs;
    m_run = mnew;

    #pragma unroll
    for (int t = 0; t < 4; ++t)
      #pragma unroll
      for (int r = 0; r < 8; ++r) o[t][r] *= alpha;

    // P^T B-fragment (same-wave LDS ops are ordered: no barrier needed)
    Frag pf;
    pf.q[0] = *(const u32x4*)&sP[w][l16][(half << 4)];
    pf.q[1] = *(const u32x4*)&sP[w][l16][(half << 4) + 8];

    // O^T += V^T @ P^T (V^T rows contiguous along keys)
    #pragma unroll
    for (int t = 0; t < 4; ++t) {
      const _Float16* vrow = Vt + ((size_t)bh * KD_ + (t << 4) + l16) * L_ + kbase;
      Frag f;
      f.q[0] = *(const u32x4*)&vrow[(half << 4)];
      f.q[1] = *(const u32x4*)&vrow[(half << 4) + 8];
      o[t] = wmma_f16(f.v, pf.v, o[t]);
    }
  }

  // normalize + write: kd is contiguous in r -> packed 8B stores
  const float inv = 1.f / l_run;
  _Float16* obase = O + ((size_t)b * L_ + q) * D_ + h * KD_ + (half << 3);
  #pragma unroll
  for (int t = 0; t < 4; ++t) {
    v4h h0 = { (_Float16)(o[t][0] * inv), (_Float16)(o[t][1] * inv),
               (_Float16)(o[t][2] * inv), (_Float16)(o[t][3] * inv) };
    v4h h1 = { (_Float16)(o[t][4] * inv), (_Float16)(o[t][5] * inv),
               (_Float16)(o[t][6] * inv), (_Float16)(o[t][7] * inv) };
    _Float16* p = obase + (t << 4);
    *(v4h*)&p[0] = h0;
    *(v4h*)&p[4] = h1;
  }
}

// ---------------- launch ----------------

extern "C" void kernel_launch(void* const* d_in, const int* in_sizes, int n_in,
                              void* d_out, int out_size, void* d_ws, size_t ws_size,
                              hipStream_t stream) {
  const float*         x    = (const float*)d_in[0];
  const unsigned char* pad  = (const unsigned char*)d_in[1];
  const float*         wqkv = (const float*)d_in[2];
  const float*         wout = (const float*)d_in[3];
  float*               out  = (float*)d_out;
  char*                ws   = (char*)d_ws;

  const size_t MB = (size_t)1 << 20;
  _Float16* xh    = (_Float16*)(ws);             // 16 MB x(f16); reused as concat
  _Float16* wqkvT = (_Float16*)(ws + 16 * MB);   //  6 MB [3072][1024]
  _Float16* woutT = (_Float16*)(ws + 22 * MB);   //  2 MB [1024][1024]
  _Float16* qB    = (_Float16*)(ws + 24 * MB);   // 16 MB [B][H][L][KD]
  _Float16* kB    = (_Float16*)(ws + 40 * MB);   // 16 MB [B][H][L][KD]
  _Float16* vT    = (_Float16*)(ws + 56 * MB);   // 16 MB [B][H][KD][L]

  const int NX = B_ * L_ * D_;                   // 8388608
  cvt_f16_kernel<<<NX / 1024, 256, 0, stream>>>(x, xh, NX);
  cvt_transpose_kernel<<<(1024 * 3072) / 256, 256, 0, stream>>>(wqkv, wqkvT, 3072);
  cvt_transpose_kernel<<<(1024 * 1024) / 256, 256, 0, stream>>>(wout, woutT, 1024);

  // qkv = x @ w_qkv  (M=8192, N=3072, K=1024), split into Q/K/V^T per head
  gemm_f16_kernel<0><<<(8192 / 128) * (3072 / 128), 256, 0, stream>>>(
      xh, wqkvT, 8192, 3072, 1024, qB, kB, vT, nullptr);

  // flash attention -> concat [B][L][D] f16 (reuses xh buffer)
  attn_kernel<<<dim3(B_ * H_, L_ / 128), 256, 0, stream>>>(qB, kB, vT, pad, xh);

  // out = concat @ w_out  (M=8192, N=1024, K=1024), f32
  gemm_f16_kernel<1><<<(8192 / 128) * (1024 / 128), 256, 0, stream>>>(
      xh, woutT, 8192, 1024, 1024, nullptr, nullptr, nullptr, out);
}
